// Qwen3_5Attention_56307021250923
// MI455X (gfx1250) — matrix operand order repaired and checked by reading, not compiled, not run on hardware
//
#include <hip/hip_runtime.h>
#include <hip/hip_bf16.h>

// ---------------- problem constants ----------------
#define HIDN 2048
#define NH   16
#define NKV  4
#define HD   256
#define ROT  64
#define SEQ  2048
#define BB   2
#define MTOT (BB * SEQ)   // 4096 token rows

// ---------------- vector types ----------------
typedef __bf16 bf16_t;
typedef __bf16 v16bf __attribute__((ext_vector_type(16)));
typedef __bf16 v8bf  __attribute__((ext_vector_type(8)));
typedef float  v8f   __attribute__((ext_vector_type(8)));
typedef unsigned int u32x4 __attribute__((ext_vector_type(4)));
typedef int i32x4 __attribute__((ext_vector_type(4)));
typedef int i32x8 __attribute__((ext_vector_type(8)));

// =====================================================================
// Kernel 0: one-shot fp32 -> bf16 conversion (hidden states are re-read
// by four projection GEMMs; converting once saves ~50 MB of HBM traffic
// and enables pure-byte async LDS staging in the GEMMs).
// =====================================================================
__global__ __launch_bounds__(256)
void convert_f32_bf16(const float* __restrict__ x, bf16_t* __restrict__ y)
{
  const size_t i = ((size_t)blockIdx.x * 256 + threadIdx.x) * 8;
  const float4 a = *(const float4*)(x + i);
  const float4 b = *(const float4*)(x + i + 4);
  v8bf o;
  o[0] = (bf16_t)a.x; o[1] = (bf16_t)a.y; o[2] = (bf16_t)a.z; o[3] = (bf16_t)a.w;
  o[4] = (bf16_t)b.x; o[5] = (bf16_t)b.y; o[6] = (bf16_t)b.z; o[7] = (bf16_t)b.w;
  *(v8bf*)(y + i) = o;
}

// =====================================================================
// Kernel 1: tiled GEMM  C[M,N] = A[M,K] * W[K,N]; A is bf16 in global
// (async-copied straight to LDS), W is fp32 (gather -> convert -> LDS
// transpose).  BM=128 BN=64 BK=32, 8 waves, 2x2 WMMA per wave.
// =====================================================================
#define BM 128
#define BN 64
#define BK 32
#define APAD 40   // padded LDS row stride in bf16 elems (80B)

__global__ __launch_bounds__(256)
void gemm_bf16A_wmma(const bf16_t* __restrict__ A, const float* __restrict__ W,
                     float* __restrict__ C, int M, int N, int K)
{
  __shared__ bf16_t As[BM * APAD];  // [m][k]
  __shared__ bf16_t Bs[BN * APAD];  // [n][k]  (W tile transposed)

  const int tid  = threadIdx.x;
  const int wave = tid >> 5;
  const int lane = tid & 31;
  const int hi   = lane >> 4;
  const int l16  = lane & 15;

  const int wm = (wave & 3) * 32;
  const int wn = (wave >> 2) * 32;
  const int bm = blockIdx.x * BM;
  const int bn = blockIdx.y * BN;

  const unsigned asLds = (unsigned)(uintptr_t)&As[0];

  const v8f vzero = {0.f,0.f,0.f,0.f,0.f,0.f,0.f,0.f};
  v8f acc[2][2];
  for (int i = 0; i < 2; ++i)
    for (int j = 0; j < 2; ++j) acc[i][j] = vzero;

  for (int k0 = 0; k0 < K; k0 += BK) {
    // ---- A tile: 128 rows x 64B, pure byte copy via async-to-LDS ----
#pragma unroll
    for (int j = 0; j < 2; ++j) {
      const int i = tid + j * 256;        // 512 x 16B chunks
      const int m = i >> 2;
      const int c = i & 3;
      const unsigned long long ga =
          (unsigned long long)(uintptr_t)(A + (size_t)(bm + m) * K + k0 + c * 8);
      const unsigned la = asLds + (unsigned)(m * 80 + c * 16);
      asm volatile("global_load_async_to_lds_b128 %0, %1, off"
                   :: "v"(la), "v"(ga) : "memory");
    }
    // ---- W tile: gather 8 loads first (MLP), then convert + transpose ----
    float wreg[8];
#pragma unroll
    for (int j = 0; j < 8; ++j) {
      const int i  = tid + j * 256;
      const int kk = i >> 6;
      const int n  = i & 63;
      wreg[j] = W[(size_t)(k0 + kk) * N + bn + n];
    }
#pragma unroll
    for (int j = 0; j < 8; ++j) {
      const int i  = tid + j * 256;
      const int kk = i >> 6;
      const int n  = i & 63;
      Bs[n * APAD + kk] = (bf16_t)wreg[j];
    }
    // prefetch next W tile (dominant traffic stream)
    if (k0 + BK < K)
      __builtin_prefetch(W + (size_t)(k0 + BK + (tid >> 6)) * N + bn + (tid & 63), 0, 1);
    asm volatile("s_wait_asynccnt 0x0" ::: "memory");
    __syncthreads();

#pragma unroll
    for (int mi = 0; mi < 2; ++mi) {
      const bf16_t* ap = &As[(wm + mi * 16 + l16) * APAD + hi * 8];
      v16bf afrag;
      *((v8bf*)&afrag)       = *(const v8bf*)ap;
      *(((v8bf*)&afrag) + 1) = *(const v8bf*)(ap + 16);
#pragma unroll
      for (int ni = 0; ni < 2; ++ni) {
        const bf16_t* bp = &Bs[(wn + ni * 16 + l16) * APAD + hi * 8];
        v16bf bfrag;
        *((v8bf*)&bfrag)       = *(const v8bf*)bp;
        *(((v8bf*)&bfrag) + 1) = *(const v8bf*)(bp + 16);
        acc[mi][ni] = __builtin_amdgcn_wmma_f32_16x16x32_bf16(
            false, afrag, false, bfrag, (short)0, acc[mi][ni], false, false);
      }
    }
    __syncthreads();
  }

#pragma unroll
  for (int mi = 0; mi < 2; ++mi)
#pragma unroll
    for (int ni = 0; ni < 2; ++ni)
#pragma unroll
      for (int r = 0; r < 8; ++r) {
        const int row = bm + wm + mi * 16 + r + 8 * hi;
        const int col = bn + wn + ni * 16 + l16;
        C[(size_t)row * N + col] = acc[mi][ni][r];
      }
}

// =====================================================================
// Kernel 2: fused gates + RMS-norm + partial RoPE.  One block per (b,s).
// V written transposed ([b,kh,d,s]) for TDM staging in attention.
// =====================================================================
__global__ __launch_bounds__(256)
void postproj_kernel(const float* __restrict__ qkv,   const float* __restrict__ kproj,
                     const float* __restrict__ vproj, const float* __restrict__ rproj,
                     const float* __restrict__ cosT,  const float* __restrict__ sinT,
                     const float* __restrict__ qnw,   const float* __restrict__ knw,
                     bf16_t* __restrict__ qout, bf16_t* __restrict__ kout,
                     bf16_t* __restrict__ vout)
{
  __shared__ float red[256];
  __shared__ float yv[256];
  __shared__ float gqk[256], gv[256];
  __shared__ float cs[ROT], ss[ROT];

  const int bs = blockIdx.x;
  const int b  = bs >> 11;
  const int s  = bs & (SEQ - 1);
  const int d  = threadIdx.x;

  gqk[d] = 1.f / (1.f + __expf(-rproj[(size_t)bs * 512 + d]));
  gv[d]  = 1.f / (1.f + __expf(-rproj[(size_t)bs * 512 + 256 + d]));
  if (d < ROT) {
    cs[d] = cosT[(size_t)bs * ROT + d];
    ss[d] = sinT[(size_t)bs * ROT + d];
  }
  __syncthreads();

  for (int h = 0; h < NH; ++h) {
    const float x = qkv[(size_t)bs * 8192 + h * 512 + d];
    red[d] = x * x;
    __syncthreads();
    for (int off = 128; off > 0; off >>= 1) {
      if (d < off) red[d] += red[d + off];
      __syncthreads();
    }
    const float inv = rsqrtf(red[0] * (1.f / 256.f) + 1e-6f);
    const float y = x * inv * (1.f + qnw[d]) * gqk[d];
    yv[d] = y;
    __syncthreads();
    float out;
    if (d < 32)      out = y * cs[d] - yv[d + 32] * ss[d];
    else if (d < 64) out = y * cs[d] + yv[d - 32] * ss[d];
    else             out = y;
    qout[(((size_t)(b * NH + h) * SEQ) + s) * HD + d] = (bf16_t)out;
    __syncthreads();
  }

  for (int h = 0; h < NKV; ++h) {
    const float x = kproj[(size_t)bs * 1024 + h * 256 + d];
    red[d] = x * x;
    __syncthreads();
    for (int off = 128; off > 0; off >>= 1) {
      if (d < off) red[d] += red[d + off];
      __syncthreads();
    }
    const float inv = rsqrtf(red[0] * (1.f / 256.f) + 1e-6f);
    const float y = x * inv * (1.f + knw[d]) * gqk[d];
    yv[d] = y;
    __syncthreads();
    float out;
    if (d < 32)      out = y * cs[d] - yv[d + 32] * ss[d];
    else if (d < 64) out = y * cs[d] + yv[d - 32] * ss[d];
    else             out = y;
    kout[(((size_t)(b * NKV + h) * SEQ) + s) * HD + d] = (bf16_t)out;

    const float xv = vproj[(size_t)bs * 1024 + h * 256 + d];
    vout[(((size_t)(b * NKV + h) * HD + d) * SEQ) + s] = (bf16_t)(xv * gv[d]);
    __syncthreads();
  }
}

// =====================================================================
// Kernel 3: flash attention (128 q-rows/block, 16 per wave).
// K tile: GLOBAL_LOAD_ASYNC_TO_LDS_B128; V tile: TDM with HW LDS pad.
// Online softmax; bf16 WMMA for Q*K^T and P*V.  Epilogue applies
// sigmoid(gate) and emits bf16 (feeds the async-A Wo GEMM).
// =====================================================================
#define KSTR 264
#define VSTR 40
#define PSTR 40

__global__ __launch_bounds__(256)
void attn_kernel(const bf16_t* __restrict__ q, const bf16_t* __restrict__ k,
                 const bf16_t* __restrict__ v, const float* __restrict__ qkv,
                 bf16_t* __restrict__ outg)
{
  __shared__ bf16_t Ks[32 * KSTR];
  __shared__ bf16_t Vs[HD * VSTR];
  __shared__ bf16_t Ps[8 * 16 * PSTR];

  const int tid  = threadIdx.x;
  const int wave = tid >> 5;
  const int lane = tid & 31;
  const int hi   = lane >> 4;
  const int l16  = lane & 15;

  const int idx  = blockIdx.x;
  const int qblk = idx & 15;
  const int hh   = (idx >> 4) & 15;
  const int b    = idx >> 8;
  const int kvh  = hh >> 2;

  const int qrow0 = qblk * 128 + wave * 16;

  v16bf qf[8];
  {
    const bf16_t* qp = q + (((size_t)(b * NH + hh) * SEQ) + qrow0 + l16) * HD;
#pragma unroll
    for (int kc = 0; kc < 8; ++kc) {
      const int dd = kc * 32 + hi * 8;
      *((v8bf*)&qf[kc])       = *(const v8bf*)(qp + dd);
      *(((v8bf*)&qf[kc]) + 1) = *(const v8bf*)(qp + dd + 16);
    }
  }

  const v8f vzero = {0.f,0.f,0.f,0.f,0.f,0.f,0.f,0.f};
  v8f o[16];
#pragma unroll
  for (int i = 0; i < 16; ++i) o[i] = vzero;
  float mrow[8], lrow[8];
#pragma unroll
  for (int r = 0; r < 8; ++r) { mrow[r] = -3.0e38f; lrow[r] = 0.f; }

  const bf16_t* kbase  = k + (size_t)(b * NKV + kvh) * SEQ * HD;
  const bf16_t* vbaseT = v + (size_t)(b * NKV + kvh) * HD * SEQ;
  const unsigned ksLds = (unsigned)(uintptr_t)&Ks[0];
  const unsigned vsLds = (unsigned)(uintptr_t)&Vs[0];
  const int ntiles = (qblk + 1) * 4;

  for (int kt = 0; kt < ntiles; ++kt) {
    const int kv0 = kt * 32;

#pragma unroll
    for (int j = 0; j < 4; ++j) {
      const int i    = tid + j * 256;
      const int sloc = i >> 5;
      const int dd   = (i & 31) * 8;
      const unsigned long long ga =
          (unsigned long long)(uintptr_t)(kbase + (size_t)(kv0 + sloc) * HD + dd);
      const unsigned la = ksLds + (unsigned)(sloc * KSTR + dd) * 2u;
      asm volatile("global_load_async_to_lds_b128 %0, %1, off"
                   :: "v"(la), "v"(ga) : "memory");
    }

    if (wave == 0) {
      const unsigned long long ga =
          (unsigned long long)(uintptr_t)vbaseT + (unsigned long long)kv0 * 2ull;
      u32x4 g0;
      g0[0] = 1u;
      g0[1] = vsLds;
      g0[2] = (unsigned)(ga & 0xffffffffu);
      g0[3] = (unsigned)((ga >> 32) & 0x01ffffffu) | (2u << 30);
      i32x8 g1;
      g1[0] = (1 << 16) | (1 << 20) | (3 << 22) | (3 << 25);
      g1[1] = (int)((2048u & 0xffffu) << 16);
      g1[2] = (int)((256u & 0xffffu) << 16);
      g1[3] = (int)(32u << 16);
      g1[4] = (int)256;
      g1[5] = (int)2048;
      g1[6] = 0;
      g1[7] = 0;
      i32x4 gz = {0, 0, 0, 0};
#if __clang_major__ >= 23
      i32x8 gz8 = {0, 0, 0, 0, 0, 0, 0, 0};
      __builtin_amdgcn_tensor_load_to_lds(g0, g1, gz, gz, gz8, 0);
#else
      __builtin_amdgcn_tensor_load_to_lds(g0, g1, gz, gz, 0);
#endif
      __builtin_amdgcn_s_wait_tensorcnt(0);
    }
    asm volatile("s_wait_asynccnt 0x0" ::: "memory");
    __syncthreads();

    if (kv0 <= qrow0 + 15) {
      v8f sacc[2];
#pragma unroll
      for (int cg = 0; cg < 2; ++cg) {
        v8f s_ = vzero;
#pragma unroll
        for (int kc = 0; kc < 8; ++kc) {
          const bf16_t* kp = &Ks[(cg * 16 + l16) * KSTR + kc * 32 + hi * 8];
          v16bf bfrag;
          *((v8bf*)&bfrag)       = *(const v8bf*)kp;
          *(((v8bf*)&bfrag) + 1) = *(const v8bf*)(kp + 16);
          s_ = __builtin_amdgcn_wmma_f32_16x16x32_bf16(
              false, qf[kc], false, bfrag, (short)0, s_, false, false);
        }
        sacc[cg] = s_;
      }

      const float scale = 0.0625f;
      float alpha[8];
#pragma unroll
      for (int r = 0; r < 8; ++r) {
        const int rowg = qrow0 + r + 8 * hi;
        float mx = -3.0e38f;
#pragma unroll
        for (int cg = 0; cg < 2; ++cg) {
          float sv = sacc[cg][r] * scale;
          const int colg = kv0 + cg * 16 + l16;
          sv = (colg > rowg) ? -1.0e9f : sv;
          sacc[cg][r] = sv;
          mx = fmaxf(mx, sv);
        }
#pragma unroll
        for (int m_ = 1; m_ < 16; m_ <<= 1)
          mx = fmaxf(mx, __shfl_xor(mx, m_, 32));
        const float mnew = fmaxf(mrow[r], mx);
        alpha[r] = __expf(mrow[r] - mnew);
        mrow[r] = mnew;
        float rs = 0.f;
#pragma unroll
        for (int cg = 0; cg < 2; ++cg) {
          const float p = __expf(sacc[cg][r] - mnew);
          Ps[(wave * 16 + r + 8 * hi) * PSTR + cg * 16 + l16] = (bf16_t)p;
          rs += p;
        }
#pragma unroll
        for (int m_ = 1; m_ < 16; m_ <<= 1)
          rs += __shfl_xor(rs, m_, 32);
        lrow[r] = lrow[r] * alpha[r] + rs;
      }
#pragma unroll
      for (int oc = 0; oc < 16; ++oc)
#pragma unroll
        for (int r = 0; r < 8; ++r)
          o[oc][r] *= alpha[r];

      v16bf pf;
      {
        const bf16_t* pp = &Ps[(wave * 16 + l16) * PSTR + hi * 8];
        *((v8bf*)&pf)       = *(const v8bf*)pp;
        *(((v8bf*)&pf) + 1) = *(const v8bf*)(pp + 16);
      }
#pragma unroll
      for (int oc = 0; oc < 16; ++oc) {
        const bf16_t* vp = &Vs[(oc * 16 + l16) * VSTR + hi * 8];
        v16bf vfrag;
        *((v8bf*)&vfrag)       = *(const v8bf*)vp;
        *(((v8bf*)&vfrag) + 1) = *(const v8bf*)(vp + 16);
        o[oc] = __builtin_amdgcn_wmma_f32_16x16x32_bf16(
            false, pf, false, vfrag, (short)0, o[oc], false, false);
      }
    }
    __syncthreads();
  }

#pragma unroll
  for (int oc = 0; oc < 16; ++oc)
#pragma unroll
    for (int r = 0; r < 8; ++r) {
      const int rowg = qrow0 + r + 8 * hi;
      const int d    = oc * 16 + l16;
      const float val = o[oc][r] / lrow[r];
      float g = qkv[((size_t)(b * SEQ + rowg)) * 8192 + hh * 512 + 256 + d];
      g = 1.f / (1.f + __expf(-g));
      outg[((size_t)(b * SEQ + rowg)) * 4096 + hh * 256 + d] = (bf16_t)(val * g);
    }
}

// =====================================================================
// Host-side orchestration
// =====================================================================
extern "C" void kernel_launch(void* const* d_in, const int* in_sizes, int n_in,
                              void* d_out, int out_size, void* d_ws, size_t ws_size,
                              hipStream_t stream) {
  const float* h    = (const float*)d_in[0];
  const float* cosT = (const float*)d_in[1];
  const float* sinT = (const float*)d_in[2];
  // d_in[3] = attention_mask (causal, reconstructed analytically in-kernel)
  const float* Wq   = (const float*)d_in[4];
  const float* Wk   = (const float*)d_in[5];
  const float* Wv   = (const float*)d_in[6];
  const float* Wo   = (const float*)d_in[7];
  const float* qnw  = (const float*)d_in[8];
  const float* knw  = (const float*)d_in[9];
  const float* Wr   = (const float*)d_in[10];

  char* ws = (char*)d_ws;
  bf16_t* h_b    = (bf16_t*)ws; ws += (size_t)MTOT * HIDN * 2;   // bf16 hidden
  float* qkv_f   = (float*)ws;  ws += (size_t)MTOT * 8192 * 4;
  float* kproj_f = (float*)ws;  ws += (size_t)MTOT * 1024 * 4;
  float* vproj_f = (float*)ws;  ws += (size_t)MTOT * 1024 * 4;
  float* rproj_f = (float*)ws;  ws += (size_t)MTOT * 512  * 4;
  bf16_t* q_b    = (bf16_t*)ws; ws += (size_t)BB * NH  * SEQ * HD * 2;
  bf16_t* k_b    = (bf16_t*)ws; ws += (size_t)BB * NKV * SEQ * HD * 2;
  bf16_t* v_b    = (bf16_t*)ws; ws += (size_t)BB * NKV * SEQ * HD * 2;  // [b,kh,d,s]
  bf16_t* attng  = (bf16_t*)ws; ws += (size_t)MTOT * 4096 * 2;   // gated attn (bf16)

  const dim3 blk(256);

  convert_f32_bf16<<<(MTOT * HIDN) / (256 * 8), blk, 0, stream>>>(h, h_b);

  gemm_bf16A_wmma<<<dim3(MTOT / BM, 512  / BN), blk, 0, stream>>>(h_b, Wr, rproj_f, MTOT, 512,  HIDN);
  gemm_bf16A_wmma<<<dim3(MTOT / BM, 8192 / BN), blk, 0, stream>>>(h_b, Wq, qkv_f,   MTOT, 8192, HIDN);
  gemm_bf16A_wmma<<<dim3(MTOT / BM, 1024 / BN), blk, 0, stream>>>(h_b, Wk, kproj_f, MTOT, 1024, HIDN);
  gemm_bf16A_wmma<<<dim3(MTOT / BM, 1024 / BN), blk, 0, stream>>>(h_b, Wv, vproj_f, MTOT, 1024, HIDN);

  postproj_kernel<<<MTOT, blk, 0, stream>>>(qkv_f, kproj_f, vproj_f, rproj_f,
                                            cosT, sinT, qnw, knw, q_b, k_b, v_b);

  attn_kernel<<<BB * NH * (SEQ / 128), blk, 0, stream>>>(q_b, k_b, v_b, qkv_f, attng);

  gemm_bf16A_wmma<<<dim3(MTOT / BM, HIDN / BN), blk, 0, stream>>>(attng, Wo, (float*)d_out,
                                                                  MTOT, HIDN, NH * HD);
}
